// RNNT_86449101733998
// MI455X (gfx1250) — compile-verified, matrix operand order checked by reading, not run on hardware
//
#include <hip/hip_runtime.h>
#include <hip/hip_bf16.h>

// ---------------------------------------------------------------------------
// RNN-T forward for MI455X (gfx1250, wave32).
//  - Input-projection GEMMs + join GEMMs: bf16 WMMA (v_wmma_f32_16x16x32_bf16),
//    fragments assembled from LDS via u32 loads + bit_cast (no scratch spills).
//  - LSTM recurrence: persistent grid, W_hh slice resident in LDS (loaded with
//    global_load_async_to_lds_b32), f32 WMMA (v_wmma_f32_16x16x4_f32) with
//    unconditional A-fragment loads (rows >= batch are don't-care), per-step h
//    exchange via async-to-LDS, cross-WG barrier via agent atomics + s_sleep.
//  - Fused join + softmax writes the 128MiB output once.
// ---------------------------------------------------------------------------

typedef __attribute__((ext_vector_type(2)))  float    v2f;
typedef __attribute__((ext_vector_type(8)))  float    v8f;
typedef __attribute__((ext_vector_type(8)))  unsigned v8u;
typedef __attribute__((ext_vector_type(16))) __bf16   v16bf;

#define B    4
#define H    512
#define G4H  2048   // 4*H
#define TENC 512
#define UDEC 128
#define VV   128
#define EMB  124

// Async global->LDS copy of one dword per lane (GLOBAL_LOAD_ASYNC_TO_LDS_B32,
// GV addressing). ldsoff = LDS byte offset (low 32 bits of flat LDS pointer).
__device__ __forceinline__ void async_g2l_b32(unsigned ldsoff, const void* gaddr) {
  asm volatile("global_load_async_to_lds_b32 %0, %1, off"
               :: "v"(ldsoff), "v"((unsigned long long)(uintptr_t)gaddr)
               : "memory");
}
__device__ __forceinline__ void wait_async0() {
  asm volatile("s_wait_asynccnt 0x0" ::: "memory");
}
__device__ __forceinline__ unsigned lds_off(const void* p) {
  return (unsigned)(uintptr_t)p;   // LDS aperture: offset lives in addr[31:0]
}

// ===========================================================================
// GEMM: out[M,N] = X[M,K] @ W[N,K]^T (+ bias0[n] + bias1[n])
// bf16 WMMA 16x16x32, macro-tile 64(M) x 128(N), 256 threads = 8 waves (2x4),
// each wave computes 32x32 (4 accumulators).  M%64==0, N%128==0 at all sites.
// ===========================================================================
template <bool WITHBIAS>
__global__ __launch_bounds__(256) void rnnt_gemm(
    const float* __restrict__ X, const float* __restrict__ W,
    const float* __restrict__ bias0, const float* __restrict__ bias1,
    float* __restrict__ out, int M, int N, int K)
{
  // [m][k] / [n][k]; row stride 34*2=68B keeps (k,k+1) pairs 4B-aligned and
  // spreads rows across banks (17-bank stride) -> conflict-free frag loads.
  __shared__ __align__(16) __bf16 As[64][34];
  __shared__ __align__(16) __bf16 Bs[128][34];

  const int tid  = threadIdx.x;
  const int lane = tid & 31;
  const int wave = tid >> 5;
  const int wm   = wave & 1;       // 2 waves along M
  const int wn   = wave >> 1;      // 4 waves along N
  const int mBase = blockIdx.x * 64;
  const int nBase = blockIdx.y * 128;
  const int hh  = lane >> 4;       // lane half
  const int l16 = lane & 15;

  v8f acc[2][2] = {};
  const int ksteps = (K + 31) >> 5;

  for (int ks = 0; ks < ksteps; ++ks) {
    const int k0 = ks << 5;
    __syncthreads();
    // stage A tile (64 x 32) as bf16, zero-pad K edge
    for (int i = tid; i < 64 * 32; i += 256) {
      int m = i >> 5, k = i & 31;
      float v = (k0 + k < K) ? X[(size_t)(mBase + m) * K + k0 + k] : 0.0f;
      As[m][k] = (__bf16)v;
    }
    // stage B tile (128 x 32) = W[n][k] as bf16
    for (int i = tid; i < 128 * 32; i += 256) {
      int n = i >> 5, k = i & 31;
      float v = (k0 + k < K) ? W[(size_t)(nBase + n) * K + k0 + k] : 0.0f;
      Bs[n][k] = (__bf16)v;
      if (k0 + 32 < K) __builtin_prefetch(&W[(size_t)(nBase + n) * K + k0 + 32 + k], 0, 1);
    }
    __syncthreads();

    // Fragment per ISA 16-bit 16x32 layout; each VGPR = one u32 LDS load
    // (pair k,k+1 packed):  v<4: k = 2v + 8*half ;  v>=4: k = 16+2(v-4)+8*half
    v16bf afrag[2], bfrag[2];
#pragma unroll
    for (int s = 0; s < 2; ++s) {
      const char* aRow = (const char*)&As[wm * 32 + s * 16 + l16][0];
      const char* bRow = (const char*)&Bs[wn * 32 + s * 16 + l16][0];
      v8u au, bu;
#pragma unroll
      for (int v = 0; v < 8; ++v) {
        int kk = ((v & 3) * 2) + ((v >> 2) * 16) + hh * 8;   // even
        au[v] = *(const unsigned*)(aRow + 2 * kk);
        bu[v] = *(const unsigned*)(bRow + 2 * kk);
      }
      afrag[s] = __builtin_bit_cast(v16bf, au);
      bfrag[s] = __builtin_bit_cast(v16bf, bu);
    }
#pragma unroll
    for (int i = 0; i < 2; ++i)
#pragma unroll
      for (int j = 0; j < 2; ++j)
        acc[i][j] = __builtin_amdgcn_wmma_f32_16x16x32_bf16(
            false, afrag[i], false, bfrag[j], (short)0, acc[i][j], false, false);
  }

  // C/D layout: lanes 0-15 -> M = r, lanes 16-31 -> M = r + 8 ; N = lane%16
#pragma unroll
  for (int i = 0; i < 2; ++i)
#pragma unroll
    for (int j = 0; j < 2; ++j) {
      int n = nBase + wn * 32 + j * 16 + l16;
      float bsum = 0.0f;
      if (WITHBIAS) bsum = bias0[n] + bias1[n];
#pragma unroll
      for (int r = 0; r < 8; ++r) {
        int m = mBase + wm * 32 + i * 16 + hh * 8 + r;
        out[(size_t)m * N + n] = acc[i][j][r] + bsum;
      }
    }
}

// ===========================================================================
// Embedding gather: X0[b*U+u, 0:124] = embedding[targets[b*U+u]][:]
// ===========================================================================
__global__ void rnnt_embed(const int* __restrict__ tgt,
                           const float* __restrict__ emb,
                           float* __restrict__ out)
{
  int row = blockIdx.x;
  int k = threadIdx.x;
  if (k < EMB) out[(size_t)row * EMB + k] = emb[(size_t)tgt[row] * EMB + k];
}

// ===========================================================================
// Persistent LSTM scan.  grid = (32, 2): 32 WGs per direction, 128 thr/4 waves.
// WG w owns hidden units [w*16, w*16+16); gate columns q*512 + w*16 + jj.
// W_hh slice (64 cols x 512 K, f32, 128 KiB) is async-loaded into LDS once and
// stays resident.  Per step: h (4x512) async-loaded to LDS, h@Whh^T via
// v_wmma_f32_16x16x4_f32 (M padded 4->16; A rows >= 4 feed only discarded D
// rows, so loads are unconditional), f32 gates/state, h published, grid-step
// barrier on a monotonic agent-scope counter.
// ===========================================================================
#define SC_NWG 32
struct ScanArgs {
  const float* pre[2];   // [B*T, 2048] per direction (includes b_ih + b_hh)
  const float* whh[2];   // [2048, 512] per direction
  float*       xout;     // [B*T, 1024]  (dir d -> features d*512 ..)
  float*       hbuf;     // [2][B][512]  h exchange
  unsigned*    ctr;      // [2] step counters (pre-zeroed)
  int          T;
  int          rev0, rev1;
};

__global__ __launch_bounds__(128) void rnnt_lstm_scan(ScanArgs a)
{
  __shared__ __align__(16) float Wt[512][64];  // [k][col]  128 KiB, resident
  __shared__ __align__(16) float hS[512][4];   // [k][b]
  __shared__ float gacc[4][64];                // [b][col]
  __shared__ float cS[4][16];                  // cell state for owned units

  const int dir = blockIdx.y;
  const int w   = blockIdx.x;
  const int tid = threadIdx.x;
  const int lane = tid & 31;
  const int wave = tid >> 5;       // 4 waves == 4 gates (16 cols each)
  const int hh   = lane >> 4;
  const int l16  = lane & 15;
  const int mb   = l16 & (B - 1);  // A-row source (don't-care for rows >= B)
  const int j0   = w * 16;
  const int T    = a.T;
  const int rev  = dir ? a.rev1 : a.rev0;

  const float* __restrict__ pre = a.pre[dir];
  const float* __restrict__ whh = a.whh[dir];
  unsigned* ctr = a.ctr + dir;
  float* hb = a.hbuf + dir * (B * H);

  // preload Whh slice via async-to-LDS: col r -> global row (r>>4)*H + j0 + (r&15)
  for (int i = tid; i < 64 * 512; i += 128) {
    int r = i >> 9, k = i & 511;
    int grow = (r >> 4) * H + j0 + (r & 15);
    async_g2l_b32(lds_off(&Wt[k][r]), &whh[(size_t)grow * H + k]);
  }
  wait_async0();
  if (tid < 64) cS[tid >> 4][tid & 15] = 0.0f;
  __syncthreads();

  unsigned target = 0;
  for (int s = 0; s < T; ++s) {
    // fetch h into LDS ([k][b])
    if (s == 0) {
      for (int i = tid; i < B * H; i += 128) hS[i >> 2][i & 3] = 0.0f;
    } else {
      for (int i = tid; i < B * H; i += 128) {
        int b = i >> 9, k = i & 511;
        async_g2l_b32(lds_off(&hS[k][b]), &hb[b * H + k]);
      }
      wait_async0();
    }
    __syncthreads();

    // h @ Whh^T for this WG's 64 columns; wave = one 16-col tile.
    // f32 WMMA 16x16x4 layout: VGPR0 K = 2*half, VGPR1 K = 2*half + 1.
    v8f acc = {};
#pragma unroll 4
    for (int ks = 0; ks < H / 4; ++ks) {
      int kk = ks * 4 + hh * 2;
      v2f av, bv;
      av.x = hS[kk][mb];               // unconditional: rows >= B are dont-care
      av.y = hS[kk + 1][mb];
      bv.x = Wt[kk][wave * 16 + l16];
      bv.y = Wt[kk + 1][wave * 16 + l16];
      acc = __builtin_amdgcn_wmma_f32_16x16x4_f32(
          false, av, false, bv, (short)0, acc, false, false);
    }
    // lanes 0-15 hold D rows m = r; only r < B are real
    if (hh == 0) {
#pragma unroll
      for (int r = 0; r < B; ++r) gacc[r][wave * 16 + l16] = acc[r];
    }
    __syncthreads();

    const int teff = rev ? (T - 1 - s) : s;
    if (tid < 64) {
      int b = tid >> 4, jj = tid & 15;
      size_t prow = ((size_t)b * T + teff) * G4H + j0;
      float gi = gacc[b][0 * 16 + jj] + pre[prow + 0 * H + jj];
      float gf = gacc[b][1 * 16 + jj] + pre[prow + 1 * H + jj];
      float gg = gacc[b][2 * 16 + jj] + pre[prow + 2 * H + jj];
      float go = gacc[b][3 * 16 + jj] + pre[prow + 3 * H + jj];
      float ig = 1.0f / (1.0f + __expf(-gi));
      float fg = 1.0f / (1.0f + __expf(-gf));
      float og = 1.0f / (1.0f + __expf(-go));
      float gt = tanhf(gg);
      float c  = fg * cS[b][jj] + ig * gt;
      float h  = og * tanhf(c);
      cS[b][jj] = c;
      hb[b * H + j0 + jj] = h;
      a.xout[((size_t)b * T + teff) * (2 * H) + dir * H + j0 + jj] = h;
      __threadfence();   // h stores agent-visible before the counter bump
    }
    __syncthreads();
    target += SC_NWG;
    if (tid == 0) {
      __hip_atomic_fetch_add(ctr, 1u, __ATOMIC_RELEASE, __HIP_MEMORY_SCOPE_AGENT);
      while (__hip_atomic_load(ctr, __ATOMIC_ACQUIRE, __HIP_MEMORY_SCOPE_AGENT) < target)
        __builtin_amdgcn_s_sleep(1);
    }
    __syncthreads();
    // every wave performs an acquire so its cache path observes the new h
    (void)__hip_atomic_load(ctr, __ATOMIC_ACQUIRE, __HIP_MEMORY_SCOPE_AGENT);
  }
}

// ===========================================================================
// Fused join + softmax: out[b,t,u,:] = softmax(encL[b,t,:] + decL[b,u,:] + bj)
// One wave32 per V=128 row (4 values/lane), shfl_xor reductions.
// ===========================================================================
__global__ __launch_bounds__(256) void rnnt_join_softmax(
    const float* __restrict__ encL, const float* __restrict__ decL,
    const float* __restrict__ bj, float* __restrict__ out)
{
  const int wave = threadIdx.x >> 5;
  const int lane = threadIdx.x & 31;
  const long row = (long)blockIdx.x * 8 + wave;    // (b*T + t)*U + u
  const int u = (int)(row & (UDEC - 1));
  const int t = (int)((row >> 7) & (TENC - 1));
  const int b = (int)(row >> 16);

  const float* er = encL + ((size_t)b * TENC + t) * VV;
  const float* dr = decL + ((size_t)b * UDEC + u) * VV;

  float x[4];
  float mx = -3.0e38f;
#pragma unroll
  for (int q = 0; q < 4; ++q) {
    int v = lane + q * 32;
    x[q] = er[v] + dr[v] + bj[v];
    mx = fmaxf(mx, x[q]);
  }
  for (int o = 16; o; o >>= 1) mx = fmaxf(mx, __shfl_xor(mx, o, 32));
  float sum = 0.0f;
#pragma unroll
  for (int q = 0; q < 4; ++q) { x[q] = __expf(x[q] - mx); sum += x[q]; }
  for (int o = 16; o; o >>= 1) sum += __shfl_xor(sum, o, 32);
  float inv = 1.0f / sum;
  float* orow = out + (size_t)row * VV;
#pragma unroll
  for (int q = 0; q < 4; ++q) orow[lane + q * 32] = x[q] * inv;
}

// ===========================================================================
// Host side.  Input order (flattened setup_inputs dict):
//  0 signals [4,512,80]  1 targets [4,128] int  2 embedding [128,124]
//  3..42  enc_params: layer l, dir d: base 3+l*8+d*4 -> Wih,Whh,bih,bhh
//  43..58 dec_params: base 43+l*8+d*4
//  59 W_j [128,1024]   60 b_j [128]           Workspace use ~56 MB.
// ===========================================================================
extern "C" void kernel_launch(void* const* d_in, const int* in_sizes, int n_in,
                              void* d_out, int out_size, void* d_ws, size_t ws_size,
                              hipStream_t stream)
{
  const float* signals   = (const float*)d_in[0];
  const int*   targets   = (const int*)d_in[1];
  const float* embedding = (const float*)d_in[2];
  const float* W_j       = (const float*)d_in[59];
  const float* b_j       = (const float*)d_in[60];
  float* out = (float*)d_out;

  auto encp = [&](int l, int d, int which) -> const float* {
    return (const float*)d_in[3 + l * 8 + d * 4 + which];
  };
  auto decp = [&](int l, int d, int which) -> const float* {
    return (const float*)d_in[43 + l * 8 + d * 4 + which];
  };

  char* wsb = (char*)d_ws;
  size_t off = 0;
  auto alloc = [&](size_t nfloats) -> float* {
    float* p = (float*)(wsb + off);
    off = (off + nfloats * sizeof(float) + 255) & ~(size_t)255;
    return p;
  };
  float* Xa   = alloc((size_t)B * TENC * 2 * H);   // 8 MB enc ping
  float* Xb   = alloc((size_t)B * TENC * 2 * H);   // 8 MB enc pong
  float* preF = alloc((size_t)B * TENC * G4H);     // 16 MB
  float* preB = alloc((size_t)B * TENC * G4H);     // 16 MB
  float* dXa  = alloc((size_t)B * UDEC * 2 * H);   // 2 MB
  float* dXb  = alloc((size_t)B * UDEC * 2 * H);   // 2 MB
  float* X0d  = alloc((size_t)B * UDEC * EMB);
  float* encL = alloc((size_t)B * TENC * VV);
  float* decL = alloc((size_t)B * UDEC * VV);
  float* hbuf = alloc(2 * B * H);
  unsigned* ctrs = (unsigned*)alloc(64);

  hipMemsetAsync(ctrs, 0, 64 * sizeof(unsigned), stream);

  // ---------------- Encoder ----------------
  const float* Xcur = signals;
  float* Xnxt = Xa;
  int Kin = 80;
  int Menc = B * TENC;                             // 2048
  for (int l = 0; l < 5; ++l) {
    dim3 g(Menc / 64, G4H / 128);
    rnnt_gemm<true><<<g, 256, 0, stream>>>(Xcur, encp(l, 0, 0), encp(l, 0, 2),
                                           encp(l, 0, 3), preF, Menc, G4H, Kin);
    rnnt_gemm<true><<<g, 256, 0, stream>>>(Xcur, encp(l, 1, 0), encp(l, 1, 2),
                                           encp(l, 1, 3), preB, Menc, G4H, Kin);
    ScanArgs sa;
    sa.pre[0] = preF; sa.pre[1] = preB;
    sa.whh[0] = encp(l, 0, 1); sa.whh[1] = encp(l, 1, 1);
    sa.xout = Xnxt; sa.hbuf = hbuf; sa.ctr = ctrs + 2 * l;
    sa.T = TENC; sa.rev0 = 0; sa.rev1 = 1;
    rnnt_lstm_scan<<<dim3(SC_NWG, 2), 128, 0, stream>>>(sa);
    Xcur = Xnxt;
    Xnxt = (Xnxt == Xa) ? Xb : Xa;
    Kin = 2 * H;
  }
  // enc logits: [2048,1024] @ W_j^T -> [2048,128]  (b_j added in softmax)
  rnnt_gemm<false><<<dim3(Menc / 64, 1), 256, 0, stream>>>(
      Xcur, W_j, nullptr, nullptr, encL, Menc, VV, 2 * H);

  // ---------------- Decoder ----------------
  rnnt_embed<<<B * UDEC, 128, 0, stream>>>(targets, embedding, X0d);
  const float* dXcur = X0d;
  float* dXnxt = dXa;
  Kin = EMB;
  int Mdec = B * UDEC;                             // 512
  for (int l = 0; l < 2; ++l) {
    dim3 g(Mdec / 64, G4H / 128);
    rnnt_gemm<true><<<g, 256, 0, stream>>>(dXcur, decp(l, 0, 0), decp(l, 0, 2),
                                           decp(l, 0, 3), preF, Mdec, G4H, Kin);
    rnnt_gemm<true><<<g, 256, 0, stream>>>(dXcur, decp(l, 1, 0), decp(l, 1, 2),
                                           decp(l, 1, 3), preB, Mdec, G4H, Kin);
    ScanArgs sa;
    sa.pre[0] = preF; sa.pre[1] = preB;
    sa.whh[0] = decp(l, 0, 1); sa.whh[1] = decp(l, 1, 1);
    sa.xout = dXnxt; sa.hbuf = hbuf; sa.ctr = ctrs + 10 + 2 * l;
    sa.T = UDEC; sa.rev0 = 0; sa.rev1 = 0;         // both directions forward
    rnnt_lstm_scan<<<dim3(SC_NWG, 2), 128, 0, stream>>>(sa);
    dXcur = dXnxt;
    dXnxt = (dXnxt == dXa) ? dXb : dXa;
    Kin = 2 * H;
  }
  rnnt_gemm<false><<<dim3(Mdec / 64, 1), 256, 0, stream>>>(
      dXcur, W_j, nullptr, nullptr, decL, Mdec, VV, 2 * H);

  // ---------------- Join + softmax ----------------
  long rows = (long)B * TENC * UDEC;               // 262144
  rnnt_join_softmax<<<(unsigned)(rows / 8), 256, 0, stream>>>(encL, decL, b_j, out);
}
